// QRNNLayer_40742059769920
// MI455X (gfx1250) — matrix operand order repaired and checked by reading, not compiled, not run on hardware
//
#include <hip/hip_runtime.h>
#include <hip/hip_bf16.h>

// ---------------------------------------------------------------------------
// QRNN fo-pool layer, MI455X (gfx1250).  B=16, D=1024, T=1024, H=1024, K=2
// Phase 1: causal conv == GEMM gates[B*T,3H] = A[B*T,2D] * Wb^T via
//          v_wmma_f32_16x16x32_bf16, double-buffered LDS, branch-free loads
//          (causal shift handled by row-shifted LDS stores), fused bias +
//          tanh/sigmoid epilogue.
// Phase 2: fo-pool scan, vectorized 4-wide, non-temporal streaming, in place
//          on d_out (z-area becomes c_seq, f-area becomes h_seq).
// ---------------------------------------------------------------------------

#define BQ 16
#define DQ 1024
#define TQ 1024
#define HQ 1024
#define NQ (3 * HQ)

#define TILE_M 128
#define TILE_N 128
#define TILE_K 32
#define LDS_STRIDE 40          // bf16 elems per row: 32 + 8 pad (80B, 16B-mult)
#define AROWS (TILE_M + 1)     // +1 row: phase-1 causal shift dump/extra row

typedef __attribute__((ext_vector_type(16))) __bf16 v16bf;
typedef __attribute__((ext_vector_type(8)))  float  v8f;
typedef __attribute__((ext_vector_type(4)))  float  f32x4;
typedef __attribute__((ext_vector_type(2)))  float  f32x2;

// Pack two f32 -> packed bf16 pair (lo in [15:0], hi in [31:16]).
__device__ __forceinline__ unsigned int pack2_bf16(float lo, float hi) {
#if __has_builtin(__builtin_amdgcn_cvt_pk_bf16_f32)
    typedef __attribute__((ext_vector_type(2))) __bf16 v2bf;
    union { v2bf v; unsigned int u; } cv;
    cv.v = __builtin_amdgcn_cvt_pk_bf16_f32(lo, hi);
    return cv.u;
#else
    unsigned int ulo = __float_as_uint(lo);
    unsigned int uhi = __float_as_uint(hi);
    ulo += 0x7FFFu + ((ulo >> 16) & 1u);   // RNE
    uhi += 0x7FFFu + ((uhi >> 16) & 1u);
    return (ulo >> 16) | (uhi & 0xFFFF0000u);
#endif
}

__device__ __forceinline__ float fast_sigmoid(float x) {
    return 1.0f / (1.0f + __expf(-x));
}

// ---------------------------------------------------------------------------
// Kernel 1: gates GEMM + activations.
// Grid (NQ/TILE_N, B*T/TILE_M), 256 threads = 8 wave32s (2 x 4 wave tiles).
// ---------------------------------------------------------------------------
__global__ __launch_bounds__(256)
void qrnn_gates_wmma(const float* __restrict__ x,      // [B, D, T]
                     const float* __restrict__ w,      // [3H, D, 2]
                     const float* __restrict__ bias,   // [3H]
                     float* __restrict__ zbuf,         // [B*T, H] tanh(z)
                     float* __restrict__ fbuf,         // [B*T, H] sig(f)
                     float* __restrict__ obuf)         // [B*T, H] sig(o)
{
    __shared__ __align__(16) unsigned short As[2][AROWS * LDS_STRIDE];
    __shared__ __align__(16) unsigned short Bs[2][TILE_N * LDS_STRIDE];

    const int tid   = threadIdx.x;
    const int lane  = tid & 31;
    const int wave  = tid >> 5;
    const int waveM = wave >> 2;      // 0..1 -> 64 rows
    const int waveN = wave & 3;       // 0..3 -> 32 cols
    const int half  = lane >> 4;
    const int l15   = lane & 15;

    const int m0 = blockIdx.y * TILE_M;       // global row = b*T + t
    const int n0 = blockIdx.x * TILE_N;       // global col = oc
    const int b  = m0 / TQ;
    const int t0 = m0 % TQ;

    const float* __restrict__ xb = x + (size_t)b * DQ * TQ;

    v8f acc[4][2];
    #pragma unroll
    for (int i = 0; i < 4; ++i)
        #pragma unroll
        for (int j = 0; j < 2; ++j)
            acc[i][j] = (v8f){0.f, 0.f, 0.f, 0.f, 0.f, 0.f, 0.f, 0.f};

    // Staging registers (global -> regs -> LDS double buffer).
    f32x2 aReg[4][2];          // [i][d of pair], each = 2 consecutive t
    f32x4 bReg[8];             // both conv taps for a d-pair
    unsigned int row0Reg = 0;  // phase-1 row 0 (x[t0-1]) packed pair

    // ---- global loads for k-tile kt into staging regs (branch-free) ----
    auto gload = [&](int kt) {
        const int p  = kt >> 5;                 // 0: tap 1 (t), 1: tap 0 (t-1)
        const int d0 = (kt & 31) * TILE_K;
        #pragma unroll
        for (int i = 0; i < 4; ++i) {
            const int linear = tid + i * 256;   // 0..1023
            const int tq2 = linear & 63;        // t pair, coalesced in lane
            const int kcp = linear >> 6;        // d pair 0..15
            const float* s0 = xb + (size_t)(d0 + kcp * 2) * TQ + t0 + 2 * tq2;
            aReg[i][0] = *(const f32x2*)(s0);
            aReg[i][1] = *(const f32x2*)(s0 + TQ);
        }
        #pragma unroll
        for (int i = 0; i < 8; ++i) {
            const int linear = tid + i * 256;   // 0..2047
            const int kcp = linear & 15;        // d pair, coalesced in lane
            const int n   = linear >> 4;        // oc 0..127
            bReg[i] = *(const f32x4*)(w + ((size_t)(n0 + n) * DQ + d0 + kcp * 2) * 2);
        }
        if (p && tid < 16) {                    // phase-1 causal row 0
            const float* s0 = xb + (size_t)(d0 + tid * 2) * TQ;
            const int   ts = (t0 > 0) ? (t0 - 1) : 0;
            const float sc = (t0 > 0) ? 1.0f : 0.0f;
            row0Reg = pack2_bf16(s0[ts] * sc, s0[TQ + ts] * sc);
        }
        // cache-warm next x k-tile (global_prefetch_b8)
        if (kt + 1 < 64)
            __builtin_prefetch(xb + (size_t)(((kt + 1) & 31) * TILE_K + (tid & 31)) * TQ + t0, 0, 1);
    };

    // ---- staged regs -> LDS buffer (bf16-pair packed b32 stores) ----
    auto lstore = [&](int kt, int buf) {
        const int p = kt >> 5;
        unsigned short* __restrict__ A = As[buf];
        #pragma unroll
        for (int i = 0; i < 4; ++i) {
            const int linear = tid + i * 256;
            const int tq2 = linear & 63;
            const int kcp = linear >> 6;
            #pragma unroll
            for (int j = 0; j < 2; ++j) {
                // causal shift: phase-1 value x[t] lands in row t-t0+1
                const int rt = 2 * tq2 + j + p;            // 0..128
                *(unsigned int*)&A[rt * LDS_STRIDE + kcp * 2] =
                    pack2_bf16(aReg[i][0][j], aReg[i][1][j]);
            }
        }
        if (p && tid < 16)
            *(unsigned int*)&A[tid * 2] = row0Reg;         // row 0
        unsigned short* __restrict__ Bsh = Bs[buf];
        #pragma unroll
        for (int i = 0; i < 8; ++i) {
            const int linear = tid + i * 256;
            const int kcp = linear & 15;
            const int n   = linear >> 4;
            // ksel = 1-p: p==0 -> taps .y/.w ; p==1 -> taps .x/.z
            const float lo = p ? bReg[i].x : bReg[i].y;
            const float hi = p ? bReg[i].z : bReg[i].w;
            *(unsigned int*)&Bsh[n * LDS_STRIDE + kcp * 2] = pack2_bf16(lo, hi);
        }
    };

    // ---- main loop: double-buffered, single barrier per k-step ----
    gload(0);
    lstore(0, 0);
    for (int kt = 0; kt < 64; ++kt) {
        const int buf = kt & 1;
        __syncthreads();                       // LDS[buf] ready, LDS[buf^1] free

        if (kt + 1 < 64) gload(kt + 1);        // overlap global with WMMA

        // fragments: A (16x32 bf16) elems 0..7 <-> K=half*8+j,
        //            elems 8..15 <-> K=16+half*8+(j-8); B same as A
        union Frag { uint4 q[2]; v16bf v; };
        v16bf afr[4], bfr[2];
        #pragma unroll
        for (int mf = 0; mf < 4; ++mf) {
            const int row = waveM * 64 + mf * 16 + l15;
            const unsigned short* base = &As[buf][row * LDS_STRIDE];
            Frag fa;
            fa.q[0] = *(const uint4*)(base + half * 8);
            fa.q[1] = *(const uint4*)(base + 16 + half * 8);
            afr[mf] = fa.v;
        }
        #pragma unroll
        for (int nf = 0; nf < 2; ++nf) {
            const int col = waveN * 32 + nf * 16 + l15;
            const unsigned short* base = &Bs[buf][col * LDS_STRIDE + half * 8];
            Frag fb;
            fb.q[0] = *(const uint4*)(base);
            fb.q[1] = *(const uint4*)(base + 16);
            bfr[nf] = fb.v;
        }

        #pragma unroll
        for (int mf = 0; mf < 4; ++mf)
            #pragma unroll
            for (int nf = 0; nf < 2; ++nf)
                acc[mf][nf] = __builtin_amdgcn_wmma_f32_16x16x32_bf16(
                    false, afr[mf], false, bfr[nf],
                    (short)0, acc[mf][nf], false, false);

        if (kt + 1 < 64) lstore(kt + 1, buf ^ 1);
    }

    // ---- epilogue: bias + activation + scatter per gate region ----
    // C/D layout: VGPR r, lanes 0-15 -> M=r, lanes 16-31 -> M=8+r; N=l15.
    #pragma unroll
    for (int mf = 0; mf < 4; ++mf) {
        #pragma unroll
        for (int nf = 0; nf < 2; ++nf) {
            const int col    = n0 + waveN * 32 + nf * 16 + l15;
            const float bv   = bias[col];
            const int region = col >> 10;      // 0:z 1:f 2:o  (H = 1024)
            const int hcol   = col & (HQ - 1);
            float* __restrict__ dst =
                (region == 0) ? zbuf : (region == 1) ? fbuf : obuf;
            #pragma unroll
            for (int r = 0; r < 8; ++r) {
                const int row = m0 + waveM * 64 + mf * 16 + half * 8 + r;
                const float g = acc[mf][nf][r] + bv;
                const float v = (region == 0) ? tanhf(g) : fast_sigmoid(g);
                dst[(size_t)row * HQ + hcol] = v;
            }
        }
    }
}

// ---------------------------------------------------------------------------
// Kernel 2: fo-pool scan, 4 h-lanes per thread, b128 non-temporal streaming,
// in place on d_out. Addresses are independent of the serial c-dependency so
// loads pipeline ahead of the recurrence.
// ---------------------------------------------------------------------------
__global__ __launch_bounds__(256)
void qrnn_scan(float* __restrict__ zc,        // in tanh(z) -> out c_seq
               float* __restrict__ fh,        // in sig(f)  -> out h_seq
               const float* __restrict__ ob)  // sig(o)
{
    const int tid = blockIdx.x * blockDim.x + threadIdx.x;  // 0 .. B*H/4-1
    const int b  = tid >> 8;                  // H/4 = 256 quads per batch
    const int h4 = (tid & 255) * 4;
    const size_t base = (size_t)b * TQ * HQ + h4;

    f32x4 c = {0.f, 0.f, 0.f, 0.f};
    #pragma unroll 2
    for (int t = 0; t < TQ; ++t) {
        const size_t idx = base + (size_t)t * HQ;
        const f32x4 z = __builtin_nontemporal_load((const f32x4*)(zc + idx));
        const f32x4 f = __builtin_nontemporal_load((const f32x4*)(fh + idx));
        const f32x4 o = __builtin_nontemporal_load((const f32x4*)(ob + idx));
        c = f * c + (1.0f - f) * z;           // fo-pooling recurrence
        __builtin_nontemporal_store(c,     (f32x4*)(zc + idx));   // c_seq
        __builtin_nontemporal_store(o * c, (f32x4*)(fh + idx));   // h_seq
    }
}

// ---------------------------------------------------------------------------
extern "C" void kernel_launch(void* const* d_in, const int* in_sizes, int n_in,
                              void* d_out, int out_size, void* d_ws, size_t ws_size,
                              hipStream_t stream)
{
    const float* x    = (const float*)d_in[0];
    const float* w    = (const float*)d_in[1];
    const float* bias = (const float*)d_in[2];

    float* out = (float*)d_out;
    float* zc  = out;                            // c_seq area [B,T,H]
    float* fh  = out + (size_t)BQ * TQ * HQ;     // h_seq area [B,T,H]
    float* ob  = (float*)d_ws;                   // o gate, B*T*H f32

    dim3 grid(NQ / TILE_N, (BQ * TQ) / TILE_M);  // (24, 128)
    qrnn_gates_wmma<<<grid, 256, 0, stream>>>(x, w, bias, zc, fh, ob);
    qrnn_scan<<<(BQ * HQ / 4) / 256, 256, 0, stream>>>(zc, fh, ob);
}
